// RVT_BlockTranspose_40003325395039
// MI455X (gfx1250) — compile-verified, matrix operand order checked by reading, not run on hardware
//
#include <hip/hip_runtime.h>
#include <math.h>

// ---------------- CDNA5 types ----------------
typedef __bf16 bf16;
typedef __attribute__((ext_vector_type(16))) __bf16 bf16x16;
typedef __attribute__((ext_vector_type(8)))  float  floatx8;
typedef __attribute__((ext_vector_type(4)))  int    v4i;

#define TOK   100352   // 8*112*112 tokens
#define CH    192
#define EH    768
#define NHD   8
#define HDM   24
#define WSZ   7
#define LW    49
#define LEPS  1e-3f
#define RSCALE 1e-5f
#define QSCALE 0.2041241452319315f   // 24^-0.5

#if __has_builtin(__builtin_amdgcn_global_load_async_to_lds_b128) && __has_builtin(__builtin_amdgcn_s_wait_asynccnt)
#define HAVE_ASYNC_LDS 1
#define AS1 __attribute__((address_space(1)))
#define AS3 __attribute__((address_space(3)))
#else
#define HAVE_ASYNC_LDS 0
#endif

union AFrag { uint4 u[2]; bf16x16 v; };

__device__ __forceinline__ floatx8 wmma_bf16(bf16x16 a, bf16x16 b, floatx8 c) {
  return __builtin_amdgcn_wmma_f32_16x16x32_bf16(false, a, false, b, (short)0, c, false, false);
}

__device__ __forceinline__ float gelu_exact(float x) {
  return 0.5f * x * (1.0f + erff(x * 0.7071067811865475f));
}

// Load a 16-bit A/B fragment (16 rows x 32 K, row-major, 2-byte elems) whose row
// base is 16B-aligned: halves [hi*8..hi*8+7] then [16+hi*8..16+hi*8+7].
__device__ __forceinline__ AFrag load_frag_rowmajor(const bf16* rowbase_k0, int hi) {
  AFrag f;
  const uint4* p = reinterpret_cast<const uint4*>(rowbase_k0 + hi * 8);
  f.u[0] = p[0];
  f.u[1] = p[2];
  return f;
}

// ---------------- one-time converts ----------------
__global__ __launch_bounds__(256) void cvt_bf16_kernel(const float* __restrict__ src,
                                                       bf16* __restrict__ dst, int n) {
  int i = blockIdx.x * 256 + threadIdx.x;
  if (i < n) dst[i] = (bf16)src[i];
}

// W[k][n] (f32, KxN) -> WT[n][k] (bf16, NxK)
__global__ __launch_bounds__(256) void transpose_cvt_kernel(const float* __restrict__ W,
                                                            bf16* __restrict__ WT,
                                                            int K, int N) {
  int i = blockIdx.x * 256 + threadIdx.x;
  if (i < K * N) {
    int k = i / N, n = i % N;
    WT[(size_t)n * K + k] = (bf16)W[i];
  }
}

// ---------------- conv (full conv pad 2) + bias + gelu, implicit GEMM ----------------
// xbf: bf16 input [8,110,110,192]; wt: bf16 [co][ (kh*3+kw)*192 + ci ]
__global__ __launch_bounds__(256) void conv_gelu_kernel(
    const bf16* __restrict__ xbf, const bf16* __restrict__ wt,
    const float* __restrict__ bias, float* __restrict__ out) {
  int wid = threadIdx.x >> 5, lane = threadIdx.x & 31;
  int tile = blockIdx.x * 8 + wid;           // (TOK/16)*(CH/16) tiles
  int m0 = (tile / 12) << 4, n0 = (tile % 12) << 4;
  int hi = lane >> 4, lm = lane & 15;
  int m = m0 + lm;
  int b  = m / (112 * 112);
  int rem = m % (112 * 112);
  int oh = rem / 112, ow = rem % 112;
  int n = n0 + lm;
  const bf16* wrow = wt + (size_t)n * (9 * CH);

  floatx8 acc = {};
  for (int kh = 0; kh < 3; ++kh) {
    int ih = oh - 2 + kh;
    for (int kw = 0; kw < 3; ++kw) {
      int iw = ow - 2 + kw;
      bool valid = (ih >= 0) && (ih < 110) && (iw >= 0) && (iw < 110);
      const bf16* ap = xbf + (size_t)((b * 110 + ih) * 110 + iw) * CH;
      const bf16* wp = wrow + (kh * 3 + kw) * CH;
      for (int c0 = 0; c0 < CH; c0 += 32) {
        AFrag a, bb;
        if (valid) {
          a = load_frag_rowmajor(ap + c0, hi);
        } else {
#pragma unroll
          for (int i = 0; i < 16; ++i) a.v[i] = (bf16)0.0f;
        }
        bb = load_frag_rowmajor(wp + c0, hi);
        acc = wmma_bf16(a.v, bb.v, acc);
      }
    }
  }
  float bn = bias[n];
#pragma unroll
  for (int r = 0; r < 8; ++r) {
    int row = m0 + r + hi * 8;
    out[(size_t)row * CH + n] = gelu_exact(acc[r] + bn);
  }
}

// ---------------- layernorm: f32 residual -> bf16 ----------------
__global__ __launch_bounds__(256) void ln_kernel(
    const float* __restrict__ x, const float* __restrict__ g,
    const float* __restrict__ bta, bf16* __restrict__ out) {
  int wid = threadIdx.x >> 5, lane = threadIdx.x & 31;
  int t = blockIdx.x * 8 + wid;
  const float* row = x + (size_t)t * CH;
  float v[6], s = 0.f;
#pragma unroll
  for (int i = 0; i < 6; ++i) { v[i] = row[lane * 6 + i]; s += v[i]; }
#pragma unroll
  for (int msk = 16; msk > 0; msk >>= 1) s += __shfl_xor(s, msk, 32);
  float mu = s * (1.0f / CH);
  float s2 = 0.f;
#pragma unroll
  for (int i = 0; i < 6; ++i) { float d = v[i] - mu; s2 += d * d; }
#pragma unroll
  for (int msk = 16; msk > 0; msk >>= 1) s2 += __shfl_xor(s2, msk, 32);
  float inv = rsqrtf(s2 * (1.0f / CH) + LEPS);
#pragma unroll
  for (int i = 0; i < 6; ++i) {
    int c = lane * 6 + i;
    out[(size_t)t * CH + c] = (bf16)((v[i] - mu) * inv * g[c] + bta[c]);
  }
}

// ---------------- generic WMMA GEMM: bf16[T,K] x bf16 WT[N,K] + bias ----------------
#define EPI_BF16 0
#define EPI_BF16_GELU 1
#define EPI_RESID 2

template <int EPI>
__global__ __launch_bounds__(256) void gemm_kernel(
    const bf16* __restrict__ A, const bf16* __restrict__ WT,
    const float* __restrict__ bias, int K, int N,
    bf16* __restrict__ outb, float* __restrict__ resid, float oscale) {
  int wid = threadIdx.x >> 5, lane = threadIdx.x & 31;
  int tile = blockIdx.x * 8 + wid;
  int nt = N >> 4;
  int m0 = (tile / nt) << 4, n0 = (tile % nt) << 4;
  int hi = lane >> 4, lm = lane & 15;
  const bf16* arow = A  + (size_t)(m0 + lm) * K;
  const bf16* brow = WT + (size_t)(n0 + lm) * K;
  int n = n0 + lm;
  floatx8 acc = {};
  for (int k0 = 0; k0 < K; k0 += 32) {
    AFrag a = load_frag_rowmajor(arow + k0, hi);
    AFrag b = load_frag_rowmajor(brow + k0, hi);
    if (k0 + 64 < K) __builtin_prefetch(brow + k0 + 64, 0, 1);
    acc = wmma_bf16(a.v, b.v, acc);
  }
  float bn = bias[n];
#pragma unroll
  for (int r = 0; r < 8; ++r) {
    int row = m0 + r + hi * 8;
    float val = acc[r] + bn;
    if (EPI == EPI_BF16) {
      outb[(size_t)row * N + n] = (bf16)(val * oscale);
    } else if (EPI == EPI_BF16_GELU) {
      outb[(size_t)row * N + n] = (bf16)gelu_exact(val);
    } else {
      size_t o = (size_t)row * N + n;
      resid[o] += RSCALE * val;
    }
  }
}

// ---------------- windowed/grid attention, one block per (window, head) ----------------
template <int MODE>  // 0 = block (7x7 windows), 1 = grid (stride-16 sampling)
__global__ __launch_bounds__(128) void attn_kernel(
    const bf16* __restrict__ Q, const bf16* __restrict__ Kb,
    const bf16* __restrict__ V, const float* __restrict__ rel,
    bf16* __restrict__ O) {
  __shared__ __align__(16) bf16 qs[64 * 32];     // q rows, padded
  __shared__ __align__(16) bf16 ks[64 * 32];     // k rows, padded
  __shared__ __align__(16) bf16 vt[32 * 64];     // V^T: [d][l], padded
  __shared__ __align__(16) float pl[64 * 64];
  __shared__ __align__(16) bf16 pb[64 * 64];

  int head = blockIdx.x & 7;
  int win  = blockIdx.x >> 3;
  int wb = win >> 8;
  int wy = (win >> 4) & 15;
  int wx = win & 15;
  int tid = threadIdx.x;

  auto token_of = [&](int l) -> int {
    int py = l / 7, px = l % 7;
    int y, x;
    if (MODE == 0) { y = wy * 7 + py;  x = wx * 7 + px; }
    else           { y = py * 16 + wy; x = px * 16 + wx; }
    return (wb * 112 + y) * 112 + x;
  };

  // stage: V transposed via direct stores; Q/K via async DMA (or direct fallback)
  for (int s = tid; s < 64 * 32; s += 128) {
    int r = s >> 5, c = s & 31;
    bool ok = (r < LW) && (c < HDM);
    size_t idx = ok ? ((size_t)token_of(r) * CH + head * HDM + c) : 0;
    vt[c * 64 + r] = ok ? V[idx] : (bf16)0.f;
#if HAVE_ASYNC_LDS
    qs[s] = (bf16)0.f;
    ks[s] = (bf16)0.f;
#else
    qs[s] = ok ? Q[idx]  : (bf16)0.f;
    ks[s] = ok ? Kb[idx] : (bf16)0.f;
#endif
  }
  __syncthreads();
#if HAVE_ASYNC_LDS
  // 49 rows * 3 16-byte chunks * 2 matrices = 294 async b128 transfers
  for (int s = tid; s < LW * 3 * 2; s += 128) {
    int mat = s & 1;
    int rem = s >> 1;
    int r = rem / 3, c0 = (rem % 3) * 8;
    const bf16* gp = (mat ? Kb : Q) + (size_t)token_of(r) * CH + head * HDM + c0;
    bf16* lp = (mat ? ks : qs) + r * 32 + c0;
    __builtin_amdgcn_global_load_async_to_lds_b128((AS1 v4i*)gp, (AS3 v4i*)lp, 0, 0);
  }
  __builtin_amdgcn_s_wait_asynccnt(0);
  __syncthreads();
#endif

  int wv = tid >> 5, lane = tid & 31;
  int hi = lane >> 4, lm = lane & 15;
  int mrow = wv << 4;

  // logits = Q * K^T   (K^T frag == A-style load of ks rows)
  AFrag aq = load_frag_rowmajor(qs + (mrow + lm) * 32, hi);
#pragma unroll
  for (int ntile = 0; ntile < 4; ++ntile) {
    AFrag bk = load_frag_rowmajor(ks + (ntile * 16 + lm) * 32, hi);
    floatx8 acc = {};
    acc = wmma_bf16(aq.v, bk.v, acc);
#pragma unroll
    for (int r = 0; r < 8; ++r) {
      int lq = mrow + r + hi * 8;
      int lk = ntile * 16 + lm;
      float val = acc[r];
      if (lk >= LW) {
        val = -1e30f;
      } else if (lq < LW) {
        int dy = lq / 7 - lk / 7 + 6;
        int dx = lq % 7 - lk % 7 + 6;
        val += rel[(dy * 13 + dx) * NHD + head];
      }
      pl[lq * 64 + lk] = val;
    }
  }
  __syncthreads();

  if (tid < 64) {
    int r = tid;
    float mx = -1e30f;
    for (int c2 = 0; c2 < 64; ++c2) mx = fmaxf(mx, pl[r * 64 + c2]);
    float sum = 0.f;
    for (int c2 = 0; c2 < 64; ++c2) sum += __expf(pl[r * 64 + c2] - mx);
    float inv = 1.f / sum;
    for (int c2 = 0; c2 < 64; ++c2)
      pb[r * 64 + c2] = (r < LW) ? (bf16)(__expf(pl[r * 64 + c2] - mx) * inv)
                                 : (bf16)0.f;
  }
  __syncthreads();

  // O = P * V   (V^T frag is a contiguous-row load from vt)
#pragma unroll
  for (int ntile = 0; ntile < 2; ++ntile) {
    floatx8 acc = {};
#pragma unroll
    for (int kc = 0; kc < 2; ++kc) {
      AFrag ap = load_frag_rowmajor(pb + (mrow + lm) * 64 + kc * 32, hi);
      AFrag bv = load_frag_rowmajor(vt + (ntile * 16 + lm) * 64 + kc * 32, hi);
      acc = wmma_bf16(ap.v, bv.v, acc);
    }
#pragma unroll
    for (int r = 0; r < 8; ++r) {
      int lq = mrow + r + hi * 8;
      int d  = ntile * 16 + lm;
      if (lq < LW && d < HDM)
        O[(size_t)token_of(lq) * CH + head * HDM + d] = (bf16)acc[r];
    }
  }
}

// ---------------- orchestration ----------------
extern "C" void kernel_launch(void* const* d_in, const int* in_sizes, int n_in,
                              void* d_out, int out_size, void* d_ws, size_t ws_size,
                              hipStream_t stream) {
  (void)in_sizes; (void)n_in; (void)out_size; (void)ws_size;
  const float* xin    = (const float*)d_in[0];
  const float* conv_w = (const float*)d_in[1];
  const float* conv_b = (const float*)d_in[2];
  const float* bln_g  = (const float*)d_in[3];
  const float* bln_b  = (const float*)d_in[4];
  const float* b_qw   = (const float*)d_in[5];
  const float* b_qb   = (const float*)d_in[6];
  const float* b_kw   = (const float*)d_in[7];
  const float* b_kb   = (const float*)d_in[8];
  const float* b_vw   = (const float*)d_in[9];
  const float* b_vb   = (const float*)d_in[10];
  const float* b_ow   = (const float*)d_in[11];
  const float* b_ob   = (const float*)d_in[12];
  const float* b_rel  = (const float*)d_in[13];
  const float* m1ln_g = (const float*)d_in[14];
  const float* m1ln_b = (const float*)d_in[15];
  const float* m1_w1  = (const float*)d_in[16];
  const float* m1_b1  = (const float*)d_in[17];
  const float* m1_w2  = (const float*)d_in[18];
  const float* m1_b2  = (const float*)d_in[19];
  const float* gln_g  = (const float*)d_in[20];
  const float* gln_b  = (const float*)d_in[21];
  const float* g_qw   = (const float*)d_in[22];
  const float* g_qb   = (const float*)d_in[23];
  const float* g_kw   = (const float*)d_in[24];
  const float* g_kb   = (const float*)d_in[25];
  const float* g_vw   = (const float*)d_in[26];
  const float* g_vb   = (const float*)d_in[27];
  const float* g_ow   = (const float*)d_in[28];
  const float* g_ob   = (const float*)d_in[29];
  const float* g_rel  = (const float*)d_in[30];

  float* X = (float*)d_out;                 // residual stream lives in d_out
  char* ws = (char*)d_ws;
  const size_t SZC  = (size_t)TOK * CH * sizeof(bf16);   // 38,535,168 B
  const size_t POOL = (size_t)TOK * EH * sizeof(bf16);   // 154,140,672 B

  bf16* LN  = (bf16*)(ws);
  bf16* Qb  = (bf16*)(ws + SZC);
  bf16* Kb  = (bf16*)(ws + 2 * SZC);
  bf16* Vb  = (bf16*)(ws + 3 * SZC);
  bf16* AO  = (bf16*)(ws + 4 * SZC);
  bf16* HID = (bf16*)(ws + SZC);            // [T,768] overlaps Q/K/V/AO (disjoint in time)
  bf16* XBF = (bf16*)(ws + SZC);            // bf16 conv input, overlaps pool (used first)

  // persistent bf16 transposed weights after the pool
  char* wp = ws + SZC + POOL;
  bf16* convWT = (bf16*)wp;              wp += (size_t)CH * 9 * CH * 2;  // [192][1728]
  bf16* bqT = (bf16*)wp; wp += (size_t)CH * CH * 2;
  bf16* bkT = (bf16*)wp; wp += (size_t)CH * CH * 2;
  bf16* bvT = (bf16*)wp; wp += (size_t)CH * CH * 2;
  bf16* boT = (bf16*)wp; wp += (size_t)CH * CH * 2;
  bf16* gqT = (bf16*)wp; wp += (size_t)CH * CH * 2;
  bf16* gkT = (bf16*)wp; wp += (size_t)CH * CH * 2;
  bf16* gvT = (bf16*)wp; wp += (size_t)CH * CH * 2;
  bf16* goT = (bf16*)wp; wp += (size_t)CH * CH * 2;
  bf16* w1T = (bf16*)wp; wp += (size_t)EH * CH * 2;   // [768][192]
  bf16* w2T = (bf16*)wp; wp += (size_t)CH * EH * 2;   // [192][768]

  const int GB192 = (TOK / 16) * (CH / 16) / 8;   // 9408
  const int GB768 = (TOK / 16) * (EH / 16) / 8;   // 37632
  const int LNB   = TOK / 8;                      // 12544
  const int ATB   = 2048 * NHD;                   // 16384
  const int XIN_N = 8 * 110 * 110 * 192;          // 18,585,600

  // one-time weight/activation conversion
  cvt_bf16_kernel<<<(XIN_N + 255) / 256, 256, 0, stream>>>(xin, XBF, XIN_N);
  transpose_cvt_kernel<<<(9 * CH * CH + 255) / 256, 256, 0, stream>>>(conv_w, convWT, 9 * CH, CH);
  transpose_cvt_kernel<<<(CH * CH + 255) / 256, 256, 0, stream>>>(b_qw, bqT, CH, CH);
  transpose_cvt_kernel<<<(CH * CH + 255) / 256, 256, 0, stream>>>(b_kw, bkT, CH, CH);
  transpose_cvt_kernel<<<(CH * CH + 255) / 256, 256, 0, stream>>>(b_vw, bvT, CH, CH);
  transpose_cvt_kernel<<<(CH * CH + 255) / 256, 256, 0, stream>>>(b_ow, boT, CH, CH);
  transpose_cvt_kernel<<<(CH * CH + 255) / 256, 256, 0, stream>>>(g_qw, gqT, CH, CH);
  transpose_cvt_kernel<<<(CH * CH + 255) / 256, 256, 0, stream>>>(g_kw, gkT, CH, CH);
  transpose_cvt_kernel<<<(CH * CH + 255) / 256, 256, 0, stream>>>(g_vw, gvT, CH, CH);
  transpose_cvt_kernel<<<(CH * CH + 255) / 256, 256, 0, stream>>>(g_ow, goT, CH, CH);
  transpose_cvt_kernel<<<(CH * EH + 255) / 256, 256, 0, stream>>>(m1_w1, w1T, CH, EH);
  transpose_cvt_kernel<<<(EH * CH + 255) / 256, 256, 0, stream>>>(m1_w2, w2T, EH, CH);

  conv_gelu_kernel<<<GB192, 256, 0, stream>>>(XBF, convWT, conv_b, X);

  auto half = [&](const float* lng, const float* lnb,
                  const bf16* qwT, const float* qb, const bf16* kwT, const float* kb2,
                  const bf16* vwT, const float* vb2, const bf16* owT, const float* ob,
                  const float* rel, int mode) {
    ln_kernel<<<LNB, 256, 0, stream>>>(X, lng, lnb, LN);
    gemm_kernel<EPI_BF16><<<GB192, 256, 0, stream>>>(LN, qwT, qb, CH, CH, Qb, nullptr, (float)QSCALE);
    gemm_kernel<EPI_BF16><<<GB192, 256, 0, stream>>>(LN, kwT, kb2, CH, CH, Kb, nullptr, 1.0f);
    gemm_kernel<EPI_BF16><<<GB192, 256, 0, stream>>>(LN, vwT, vb2, CH, CH, Vb, nullptr, 1.0f);
    if (mode == 0) attn_kernel<0><<<ATB, 128, 0, stream>>>(Qb, Kb, Vb, rel, AO);
    else           attn_kernel<1><<<ATB, 128, 0, stream>>>(Qb, Kb, Vb, rel, AO);
    gemm_kernel<EPI_RESID><<<GB192, 256, 0, stream>>>(AO, owT, ob, CH, CH, nullptr, X, 1.0f);
    // FFN (mlp1 reused for both FFN sub-blocks per reference)
    ln_kernel<<<LNB, 256, 0, stream>>>(X, m1ln_g, m1ln_b, LN);
    gemm_kernel<EPI_BF16_GELU><<<GB768, 256, 0, stream>>>(LN, w1T, m1_b1, CH, EH, HID, nullptr, 1.0f);
    gemm_kernel<EPI_RESID><<<GB192, 256, 0, stream>>>(HID, w2T, m1_b2, EH, CH, nullptr, X, 1.0f);
  };

  half(bln_g, bln_b, bqT, b_qb, bkT, b_kb, bvT, b_vb, boT, b_ob, b_rel, 0);
  half(gln_g, gln_b, gqT, g_qb, gkT, g_kb, gvT, g_vb, goT, g_ob, g_rel, 1);
}
